// DVAE_25769803776288
// MI455X (gfx1250) — matrix operand order repaired and checked by reading, not compiled, not run on hardware
//
#include <hip/hip_runtime.h>
#include <hip/hip_bf16.h>
#include <stdint.h>

// Problem constants
#define B_    2048
#define MAXN_ 20
#define HS_   501
#define NZ_   56
#define KP_   512           // HS padded to 16 WMMA k-steps of 32
#define VS_   (HS_ + MAXN_) // 521, row stride of Wg/Wm
#define KLDS_ 520           // LDS row stride (bf16): 1040B = 4-bank rotation/row -> conflict-free

typedef __bf16 bf16;
typedef __attribute__((ext_vector_type(16))) __bf16 v16bf;
typedef __attribute__((ext_vector_type(8)))  __bf16 v8bf;
typedef __attribute__((ext_vector_type(8)))  float  v8f;

__device__ __forceinline__ float sigmoidf_(float x) { return 1.0f / (1.0f + __expf(-x)); }

// ---- WMMA bf16 fragment loaders (wave32; contiguous k order, same in A and B) ----
// A (16x32, bf16): lane l -> row M = l&15; K = {16*kh..16*kh+15}, kh = l>>4.
__device__ __forceinline__ v16bf load_a_frag(const bf16* __restrict__ rowbase, int lane) {
  int kh = (lane >> 4) & 1;
  v8bf a0 = *(const v8bf*)(rowbase + 16 * kh);
  v8bf a1 = *(const v8bf*)(rowbase + 16 * kh + 8);
  return __builtin_shufflevector(a0, a1, 0,1,2,3,4,5,6,7,8,9,10,11,12,13,14,15);
}
// 16 contiguous bf16 -> B fragment lane chunk (row = N-col, 16 contiguous K)
__device__ __forceinline__ v16bf load_frag16(const bf16* p) {
  v8bf b0 = *(const v8bf*)(p);
  v8bf b1 = *(const v8bf*)(p + 8);
  return __builtin_shufflevector(b0, b1, 0,1,2,3,4,5,6,7,8,9,10,11,12,13,14,15);
}

// ---- Weight packing (f32 -> bf16, zero-padded) ----
// W_hh (1503,501) -> 3 gate slabs [512][512]
__global__ __launch_bounds__(256) void k_pack_whh(const float* __restrict__ W, bf16* __restrict__ out) {
  int idx = blockIdx.x * 256 + threadIdx.x;       // 3*512*512
  int g = idx >> 18, r = (idx >> 9) & 511, k = idx & 511;
  float v = (r < HS_ && k < HS_) ? W[(size_t)(g * HS_ + r) * HS_ + k] : 0.f;
  out[idx] = (bf16)v;
}
// Wg/Wm (501,521) -> square slab [512][512] using first 501 columns
__global__ __launch_bounds__(256) void k_pack_sq(const float* __restrict__ W, bf16* __restrict__ out) {
  int idx = blockIdx.x * 256 + threadIdx.x;       // 512*512
  int r = idx >> 9, k = idx & 511;
  float v = (r < HS_ && k < HS_) ? W[(size_t)r * VS_ + k] : 0.f;
  out[idx] = (bf16)v;
}
// W1,W2 (56,501) -> [128][512]; rows 0..55 = W1, rows 64..119 = W2
__global__ __launch_bounds__(256) void k_pack_w12(const float* __restrict__ W1, const float* __restrict__ W2,
                                                  bf16* __restrict__ out) {
  int idx = blockIdx.x * 256 + threadIdx.x;       // 128*512
  int r = idx >> 9, k = idx & 511;
  float v = 0.f;
  if (k < HS_) {
    if (r < NZ_)                      v = W1[(size_t)r * HS_ + k];
    else if (r >= 64 && r < 64 + NZ_) v = W2[(size_t)(r - 64) * HS_ + k];
  }
  out[idx] = (bf16)v;
}

// ---- Step kernel 1: Hin[b,:] = sum_{u<v} adj[b,u,v] * gated[b,u,:] ----
__global__ __launch_bounds__(256) void k_aggregate(const bf16* __restrict__ gated,
                                                   const float* __restrict__ adj, int v,
                                                   float* __restrict__ HinF, bf16* __restrict__ HinB) {
  int tid = blockIdx.x * 256 + threadIdx.x;       // 2048 * 64
  int b = tid >> 6;
  int jc = (tid & 63) * 8;
  float acc[8];
  #pragma unroll
  for (int i = 0; i < 8; ++i) acc[i] = 0.f;
  for (int u = 0; u < v; ++u) {
    float m = adj[(size_t)(b * MAXN_ + u) * MAXN_ + v];
    if (m != 0.f) {
      v8bf g = *(const v8bf*)(gated + ((size_t)b * MAXN_ + u) * KP_ + jc);
      #pragma unroll
      for (int i = 0; i < 8; ++i) acc[i] += m * (float)g[i];
    }
  }
  v8bf hb;
  #pragma unroll
  for (int i = 0; i < 8; ++i) {
    HinF[(size_t)b * KP_ + jc + i] = acc[i];
    hb[i] = (bf16)acc[i];
  }
  *(v8bf*)(HinB + (size_t)b * KP_ + jc) = hb;
}

// ---- Step kernel 2: gh = Hin @ W_hh^T (3 WMMA chains, B staged in LDS) + GRU epilogue -> Hv ----
// Block: 128 M-rows x 16 N-cols; 8 waves share the LDS-resident B tile (3 gates x 16 x 512).
__global__ __launch_bounds__(256) void k_gru(const bf16* __restrict__ HinB, const float* __restrict__ HinF,
                                             const bf16* __restrict__ WhhB,
                                             const int* __restrict__ types, const float* __restrict__ feats,
                                             const float* __restrict__ W_ih, const float* __restrict__ b_ih,
                                             const float* __restrict__ b_hh, int v, bf16* __restrict__ Hv) {
  __shared__ bf16 Bs[3 * 16 * KLDS_];             // ~49 KB
  int mBlk = blockIdx.x >> 5;                     // 0..15
  int nT   = blockIdx.x & 31;                     // 0..31
  // Cooperative stage: 3 gates x 16 N-rows x 512 K, 8 bf16 per chunk
  for (int c = threadIdx.x; c < 3 * 16 * 64; c += 256) {
    int g = c >> 10;                              // /(16*64)
    int rem = c & 1023;
    int r = rem >> 6;                             // 0..15
    int k8 = (rem & 63) * 8;
    *(v8bf*)&Bs[(g * 16 + r) * KLDS_ + k8] =
        *(const v8bf*)&WhhB[((size_t)g * KP_ + nT * 16 + r) * KP_ + k8];
  }
  __syncthreads();

  int wave = threadIdx.x >> 5, lane = threadIdx.x & 31;
  int mT = mBlk * 8 + wave;                       // 0..127
  int kh = (lane >> 4) & 1;
  v8f accR = {}, accZ = {}, accN = {};
  const bf16* arow = HinB + (size_t)(mT * 16 + (lane & 15)) * KP_;
  const bf16* brow = &Bs[(lane & 15) * KLDS_ + 16 * kh];
  #pragma unroll 4
  for (int kt = 0; kt < 16; ++kt) {
    v16bf a  = load_a_frag(arow + kt * 32, lane);
    const bf16* p = brow + kt * 32;
    v16bf br = load_frag16(p);
    v16bf bz = load_frag16(p + 16 * KLDS_);
    v16bf bn = load_frag16(p + 32 * KLDS_);
    accR = __builtin_amdgcn_wmma_f32_16x16x32_bf16(false, a, false, br, (short)0, accR, false, false);
    accZ = __builtin_amdgcn_wmma_f32_16x16x32_bf16(false, a, false, bz, (short)0, accZ, false, false);
    accN = __builtin_amdgcn_wmma_f32_16x16x32_bf16(false, a, false, bn, (short)0, accN, false, false);
  }
  int n = nT * 16 + (lane & 15);
  bool valid = (n < HS_);
  int nc = valid ? n : 0;
  const float* wr = W_ih + (size_t)nc * 47;       // gi row gathers: x is one-hot(type)+feat
  const float* wz = W_ih + (size_t)(HS_ + nc) * 47;
  const float* wn = W_ih + (size_t)(2 * HS_ + nc) * 47;
  float bir = b_ih[nc], biz = b_ih[HS_ + nc], bin = b_ih[2 * HS_ + nc];
  float bhr = b_hh[nc], bhz = b_hh[HS_ + nc], bhn = b_hh[2 * HS_ + nc];
  #pragma unroll
  for (int i = 0; i < 8; ++i) {
    int m = mT * 16 + 8 * kh + i;                 // C/D layout: VGPR i -> M = 8*kh + i
    float hv = 0.f;
    if (valid) {
      int   t = types[m * MAXN_ + v];
      float f = feats[m * MAXN_ + v];
      float gr = wr[t] + f * wr[46] + bir + accR[i] + bhr;
      float gz = wz[t] + f * wz[46] + biz + accZ[i] + bhz;
      float r = sigmoidf_(gr);
      float z = sigmoidf_(gz);
      float nn = tanhf(wn[t] + f * wn[46] + bin + r * (accN[i] + bhn));
      float h = HinF[(size_t)m * KP_ + n];
      hv = (1.f - z) * nn + z * h;
    }
    Hv[(size_t)m * KP_ + n] = (bf16)hv;
  }
}

// ---- Step kernel 3: gated[b,v,:] = sigmoid(Hv@Wg'^T + Wg[:,501+v] + bg) * (Hv@Wm'^T + Wm[:,501+v]) ----
__global__ __launch_bounds__(256) void k_gated(const bf16* __restrict__ Hv,
                                               const bf16* __restrict__ WgB, const bf16* __restrict__ WmB,
                                               const float* __restrict__ Wg, const float* __restrict__ bg,
                                               const float* __restrict__ Wm, int v, bf16* __restrict__ gated) {
  __shared__ bf16 Bs[2 * 16 * KLDS_];             // ~33 KB
  int mBlk = blockIdx.x >> 5;
  int nT   = blockIdx.x & 31;
  for (int c = threadIdx.x; c < 2 * 16 * 64; c += 256) {
    int g = c >> 10;
    int rem = c & 1023;
    int r = rem >> 6;
    int k8 = (rem & 63) * 8;
    const bf16* src = (g == 0) ? WgB : WmB;
    *(v8bf*)&Bs[(g * 16 + r) * KLDS_ + k8] =
        *(const v8bf*)&src[((size_t)nT * 16 + r) * KP_ + k8];
  }
  __syncthreads();

  int wave = threadIdx.x >> 5, lane = threadIdx.x & 31;
  int mT = mBlk * 8 + wave;
  int kh = (lane >> 4) & 1;
  v8f aG = {}, aM = {};
  const bf16* arow = Hv + (size_t)(mT * 16 + (lane & 15)) * KP_;
  const bf16* brow = &Bs[(lane & 15) * KLDS_ + 16 * kh];
  #pragma unroll 4
  for (int kt = 0; kt < 16; ++kt) {
    v16bf a  = load_a_frag(arow + kt * 32, lane);
    const bf16* p = brow + kt * 32;
    v16bf bG = load_frag16(p);
    v16bf bM = load_frag16(p + 16 * KLDS_);
    aG = __builtin_amdgcn_wmma_f32_16x16x32_bf16(false, a, false, bG, (short)0, aG, false, false);
    aM = __builtin_amdgcn_wmma_f32_16x16x32_bf16(false, a, false, bM, (short)0, aM, false, false);
  }
  int n = nT * 16 + (lane & 15);
  bool valid = (n < HS_);
  int nc = valid ? n : 0;
  float wgc = Wg[(size_t)nc * VS_ + HS_ + v] + bg[nc]; // identity block of Hcat -> column select
  float wmc = Wm[(size_t)nc * VS_ + HS_ + v];
  #pragma unroll
  for (int i = 0; i < 8; ++i) {
    int m = mT * 16 + 8 * kh + i;
    float o = 0.f;
    if (valid) o = sigmoidf_(aG[i] + wgc) * (aM[i] + wmc);
    gated[((size_t)m * MAXN_ + v) * KP_ + n] = (bf16)o;
  }
}

// ---- Final heads: mu = Hg@W1^T + b1, logvar = Hg@W2^T + b2 (both in one N=128 GEMM) ----
__global__ __launch_bounds__(256) void k_final(const bf16* __restrict__ Hv, const bf16* __restrict__ W12B,
                                               const float* __restrict__ b1, const float* __restrict__ b2,
                                               float* __restrict__ out) {
  int wave = threadIdx.x >> 5, lane = threadIdx.x & 31;
  int mT = blockIdx.x;                            // 128
  int nT = wave;                                  // 0..7
  int kh = (lane >> 4) & 1;
  v8f acc = {};
  const bf16* arow = Hv + (size_t)(mT * 16 + (lane & 15)) * KP_;
  #pragma unroll 4
  for (int kt = 0; kt < 16; ++kt) {
    v16bf a = load_a_frag(arow + kt * 32, lane);
    v16bf b = load_frag16(W12B + (size_t)(nT * 16 + (lane & 15)) * KP_ + kt * 32 + 16 * kh);
    acc = __builtin_amdgcn_wmma_f32_16x16x32_bf16(false, a, false, b, (short)0, acc, false, false);
  }
  int n = nT * 16 + (lane & 15);
  #pragma unroll
  for (int i = 0; i < 8; ++i) {
    int m = mT * 16 + 8 * kh + i;
    float v = acc[i];
    if (n < NZ_)                       out[(size_t)m * NZ_ + n] = v + b1[n];
    else if (n >= 64 && n < 64 + NZ_)  out[(size_t)B_ * NZ_ + (size_t)m * NZ_ + (n - 64)] = v + b2[n - 64];
  }
}

extern "C" void kernel_launch(void* const* d_in, const int* in_sizes, int n_in,
                              void* d_out, int out_size, void* d_ws, size_t ws_size,
                              hipStream_t stream) {
  const int*   types = (const int*)  d_in[0];
  const float* feats = (const float*)d_in[1];
  const float* adj   = (const float*)d_in[2];
  const float* Wg    = (const float*)d_in[3];
  const float* bg    = (const float*)d_in[4];
  const float* Wm    = (const float*)d_in[5];
  const float* W_ih  = (const float*)d_in[6];
  const float* b_ih  = (const float*)d_in[7];
  const float* W_hh  = (const float*)d_in[8];
  const float* b_hh  = (const float*)d_in[9];
  const float* W1    = (const float*)d_in[10];
  const float* b1    = (const float*)d_in[11];
  const float* W2    = (const float*)d_in[12];
  const float* b2    = (const float*)d_in[13];
  float* out = (float*)d_out;

  // Workspace carve-out (~53 MB total, L2-resident on MI455X)
  char* ws = (char*)d_ws;
  size_t off = 0;
  auto carve = [&](size_t bytes) { void* p = ws + off; off = (off + bytes + 255) & ~(size_t)255; return p; };
  bf16* whhB  = (bf16*)carve((size_t)3 * KP_ * KP_ * 2);       // 1.5 MB
  bf16* wgB   = (bf16*)carve((size_t)KP_ * KP_ * 2);           // 0.5 MB
  bf16* wmB   = (bf16*)carve((size_t)KP_ * KP_ * 2);           // 0.5 MB
  bf16* w12B  = (bf16*)carve((size_t)128 * KP_ * 2);           // 128 KB
  float* hinF = (float*)carve((size_t)B_ * KP_ * 4);           // 4 MB
  bf16* hinB  = (bf16*)carve((size_t)B_ * KP_ * 2);            // 2 MB
  bf16* hv    = (bf16*)carve((size_t)B_ * KP_ * 2);            // 2 MB
  bf16* gated = (bf16*)carve((size_t)B_ * MAXN_ * KP_ * 2);    // 42 MB

  k_pack_whh<<<3072, 256, 0, stream>>>(W_hh, whhB);
  k_pack_sq <<<1024, 256, 0, stream>>>(Wg, wgB);
  k_pack_sq <<<1024, 256, 0, stream>>>(Wm, wmB);
  k_pack_w12<<<256,  256, 0, stream>>>(W1, W2, w12B);

  for (int v = 0; v < MAXN_; ++v) {
    k_aggregate<<<512, 256, 0, stream>>>(gated, adj, v, hinF, hinB);
    k_gru<<<512, 256, 0, stream>>>(hinB, hinF, whhB, types, feats, W_ih, b_ih, b_hh, v, hv);
    if (v < MAXN_ - 1)  // gated[.,19] is never consumed (strict upper-tri adjacency)
      k_gated<<<512, 256, 0, stream>>>(hv, wgB, wmB, Wg, bg, Wm, v, gated);
  }
  k_final<<<128, 256, 0, stream>>>(hv, w12B, b1, b2, out);
}
